// AttentionPairBias_20306605375689
// MI455X (gfx1250) — matrix operand order repaired and checked by reading, not compiled, not run on hardware
//
#include <hip/hip_runtime.h>
#include <hip/hip_bf16.h>

typedef __attribute__((ext_vector_type(16))) __bf16 v16bf;
typedef __attribute__((ext_vector_type(8)))  float  v8f;
typedef __attribute__((ext_vector_type(2)))  __bf16 bf16x2;

#define NN_   1024
#define CS_   768
#define CZ_   128
#define H_    16
#define HD_   48
#define SPLIT_ 4          // attention split-K over j
#define JCHUNK_ (NN_ / SPLIT_)

// ---------------------------------------------------------------------------
// WMMA helper
// ---------------------------------------------------------------------------
__device__ inline v8f wmma_bf16(v16bf a, v16bf b, v8f c) {
  return __builtin_amdgcn_wmma_f32_16x16x32_bf16(
      /*neg_a=*/false, a, /*neg_b=*/false, b,
      /*c_mod=*/(short)0, c, /*reuse_a=*/false, /*reuse_b=*/false);
}

// A fragment: 16x32 (MxK), row-major source, lane L: M=L&15, half=L>>4
// VGPR j holds K pair: k = (j&3)*2 + (j>>2)*16 + half*8  (pairs contiguous)
__device__ inline v16bf load_a_frag(const __bf16* __restrict__ src, int ld,
                                    int row0, int col0, int kvalid) {
  const int lane = threadIdx.x & 31;
  const int m = lane & 15, half = lane >> 4;
  const __bf16* rowp = src + (size_t)(row0 + m) * ld + col0;
  v16bf a;
#pragma unroll
  for (int j = 0; j < 8; ++j) {
    int k = ((j & 3) << 1) + ((j >> 2) << 4) + (half << 3);
    __bf16 e0 = (__bf16)0.0f, e1 = (__bf16)0.0f;
    if (k < kvalid) {
      bf16x2 p = *(const bf16x2*)(rowp + k);
      e0 = p[0]; e1 = p[1];
    }
    a[2 * j] = e0; a[2 * j + 1] = e1;
  }
  return a;
}

// B fragment: 32x16 (KxN). Element (kk,n) at src[(nbase+n)*ld + kbase + kk]
// (K contiguous in memory per column n). lane L: N=L&15, half=L>>4,
// VGPR j holds K pair k = (j&3)*2 + (j>>2)*16 + half*8.
__device__ inline v16bf load_b_frag(const __bf16* __restrict__ src, int ld,
                                    int nbase, int kbase, int kvalid) {
  const int lane = threadIdx.x & 31;
  const int n = lane & 15, half = lane >> 4;
  const __bf16* rowp = src + (size_t)(nbase + n) * ld + kbase;
  v16bf b;
#pragma unroll
  for (int j = 0; j < 8; ++j) {
    int k = ((j & 3) << 1) + ((j >> 2) << 4) + (half << 3);
    __bf16 e0 = (__bf16)0.0f, e1 = (__bf16)0.0f;
    if (k < kvalid) {
      bf16x2 p = *(const bf16x2*)(rowp + k);
      e0 = p[0]; e1 = p[1];
    }
    b[2 * j] = e0; b[2 * j + 1] = e1;
  }
  return b;
}

// ---------------------------------------------------------------------------
// fp32 -> bf16 conversion
// ---------------------------------------------------------------------------
__global__ void __launch_bounds__(256)
cvt_f32_bf16(const float* __restrict__ src, __bf16* __restrict__ dst, int n) {
  int i = blockIdx.x * 256 + threadIdx.x;
  if (i < n) dst[i] = (__bf16)src[i];
}

// ---------------------------------------------------------------------------
// LayerNorm over s rows (C_S = 768), bf16 output
// ---------------------------------------------------------------------------
__global__ void __launch_bounds__(256)
ln_s_kernel(const float* __restrict__ s, const float* __restrict__ w,
            const float* __restrict__ b, __bf16* __restrict__ out) {
  __shared__ float red[256];
  const int row = blockIdx.x, tid = threadIdx.x;
  const float* sp = s + (size_t)row * CS_;
  float x[3], s1 = 0.f, s2 = 0.f;
#pragma unroll
  for (int t = 0; t < 3; ++t) {
    x[t] = sp[tid + 256 * t];
    s1 += x[t]; s2 += x[t] * x[t];
  }
  red[tid] = s1; __syncthreads();
  for (int off = 128; off > 0; off >>= 1) {
    if (tid < off) red[tid] += red[tid + off];
    __syncthreads();
  }
  float mu = red[0] * (1.f / CS_);
  __syncthreads();
  red[tid] = s2; __syncthreads();
  for (int off = 128; off > 0; off >>= 1) {
    if (tid < off) red[tid] += red[tid + off];
    __syncthreads();
  }
  float var = red[0] * (1.f / CS_) - mu * mu;
  float rs = rsqrtf(var + 1e-5f);
#pragma unroll
  for (int t = 0; t < 3; ++t) {
    int c = tid + 256 * t;
    out[(size_t)row * CS_ + c] = (__bf16)((x[t] - mu) * rs * w[c] + b[c]);
  }
}

// ---------------------------------------------------------------------------
// GEMM  Y = X @ W^T (+bias)(+sigmoid). X: MxK bf16 row-major, W: NoutxK bf16.
// One 16x16 output tile per wave, K looped in steps of 32 via WMMA.
// flags: 1=bias, 2=sigmoid, 4=output f32 (Yf), 8=output bf16 transposed
// ---------------------------------------------------------------------------
__global__ void __launch_bounds__(128)
gemm_xwT(const __bf16* __restrict__ X, const __bf16* __restrict__ W,
         const float* __restrict__ bias, float* __restrict__ Yf,
         __bf16* __restrict__ Yb, int M, int Nout, int K, int flags) {
  const int wave = threadIdx.x >> 5, lane = threadIdx.x & 31;
  const int tile = blockIdx.x * 4 + wave;
  const int nt = Nout >> 4;
  if (tile >= (M >> 4) * nt) return;
  const int m0 = (tile / nt) << 4;
  const int n0 = (tile % nt) << 4;
  const int n = lane & 15, half = lane >> 4;

  v8f acc = {};
  for (int k0 = 0; k0 < K; k0 += 32) {
    v16bf a = load_a_frag(X, K, m0, k0, 32);
    v16bf b = load_b_frag(W, K, n0, k0, 32);
    acc = wmma_bf16(a, b, acc);
  }
#pragma unroll
  for (int r = 0; r < 8; ++r) {
    int mm = r + half * 8;
    float y = acc[r];
    if (flags & 1) y += bias[n0 + n];
    if (flags & 2) y = 1.f / (1.f + __expf(-y));
    if (flags & 4)      Yf[(size_t)(m0 + mm) * Nout + n0 + n] = y;
    else if (flags & 8) Yb[(size_t)(n0 + n) * M + m0 + mm] = (__bf16)y;  // transposed
    else                Yb[(size_t)(m0 + mm) * Nout + n0 + n] = (__bf16)y;
  }
}

// ---------------------------------------------------------------------------
// Pair bias: zb[h,i,j] = sum_c LN(z[i,j,:])[c] * Wz[h,c]
// One wave handles 64 (i,j) pairs (4 WMMA tiles of 16 pairs, K=128 in 4 chunks)
// ---------------------------------------------------------------------------
__global__ void __launch_bounds__(128)
zb_kernel(const float* __restrict__ z, const float* __restrict__ lnw,
          const float* __restrict__ lnb, const __bf16* __restrict__ Wzb,
          float* __restrict__ zb) {
  __shared__ float st[4][64 * 16];
  const int wave = threadIdx.x >> 5, lane = threadIdx.x & 31;
  const int job = blockIdx.x * 4 + wave;
  const int i = job >> 4;
  const int j0 = (job & 15) << 6;
  const int m = lane & 15, half = lane >> 4;

  // Resident B fragments of Wz^T (16 heads x 128 channels)
  v16bf bw0 = load_b_frag(Wzb, CZ_, 0, 0, 32);
  v16bf bw1 = load_b_frag(Wzb, CZ_, 0, 32, 32);
  v16bf bw2 = load_b_frag(Wzb, CZ_, 0, 64, 32);
  v16bf bw3 = load_b_frag(Wzb, CZ_, 0, 96, 32);
  float* sm = st[wave];

  for (int t = 0; t < 4; ++t) {
    const int jrow = j0 + t * 16 + m;
    const float* zp = z + ((size_t)i * NN_ + jrow) * CZ_;
    // keep the 512MB z stream ahead of the LN/WMMA pipeline
    if (t < 3) __builtin_prefetch(zp + 16 * CZ_, 0, 1);
    // pass 1: this lane owns c in {f*32 + half*8 + [0..8) U [16..24)}  (64 vals)
    float s1 = 0.f, s2 = 0.f;
#pragma unroll
    for (int f = 0; f < 4; ++f) {
      int base = f * 32 + half * 8;
#pragma unroll
      for (int cc = 0; cc < 2; ++cc) {
        float4 a4 = *(const float4*)(zp + base + cc * 16);
        float4 b4 = *(const float4*)(zp + base + cc * 16 + 4);
        s1 += a4.x + a4.y + a4.z + a4.w + b4.x + b4.y + b4.z + b4.w;
        s2 += a4.x * a4.x + a4.y * a4.y + a4.z * a4.z + a4.w * a4.w +
              b4.x * b4.x + b4.y * b4.y + b4.z * b4.z + b4.w * b4.w;
      }
    }
    s1 += __shfl_xor(s1, 16, 32);
    s2 += __shfl_xor(s2, 16, 32);
    float mu = s1 * (1.f / CZ_);
    float var = s2 * (1.f / CZ_) - mu * mu;
    float rs = rsqrtf(var + 1e-5f);
    // pass 2: normalize straight into A fragments (L0-hot reload)
    v16bf af[4];
#pragma unroll
    for (int f = 0; f < 4; ++f) {
#pragma unroll
      for (int jj = 0; jj < 8; ++jj) {
        int kk = ((jj & 3) << 1) + ((jj >> 2) << 4) + (half << 3);
        int c = f * 32 + kk;
        af[f][2 * jj]     = (__bf16)((zp[c]     - mu) * rs * lnw[c]     + lnb[c]);
        af[f][2 * jj + 1] = (__bf16)((zp[c + 1] - mu) * rs * lnw[c + 1] + lnb[c + 1]);
      }
    }
    v8f c8 = {};
    c8 = wmma_bf16(af[0], bw0, c8);
    c8 = wmma_bf16(af[1], bw1, c8);
    c8 = wmma_bf16(af[2], bw2, c8);
    c8 = wmma_bf16(af[3], bw3, c8);
#pragma unroll
    for (int r = 0; r < 8; ++r)
      sm[(t * 16 + r + half * 8) * 16 + m] = c8[r];  // (pair, head)
  }
  // coalesced h-major store (per-wave LDS region; LDS is in-order per wave)
  for (int idx = lane; idx < 64 * 16; idx += 32) {
    int hh = idx >> 6, jl = idx & 63;
    zb[(size_t)hh * NN_ * NN_ + (size_t)i * NN_ + j0 + jl] = sm[jl * 16 + hh];
  }
}

// ---------------------------------------------------------------------------
// Flash attention, split-K over j for occupancy. One wave per
// (head, 16-row block, j-split). Emits unnormalized partial O + (m, l).
// ---------------------------------------------------------------------------
__global__ void __launch_bounds__(128)
attn_partial_kernel(const __bf16* __restrict__ q, const __bf16* __restrict__ k,
                    const __bf16* __restrict__ vT, const float* __restrict__ zb,
                    const float* __restrict__ mask, float* __restrict__ po,
                    float2* __restrict__ pml) {
  __shared__ __bf16 lds_p[4][16 * 32];
  const int wave = threadIdx.x >> 5, lane = threadIdx.x & 31;
  const int job = blockIdx.x * 4 + wave;       // SPLIT_*1024 jobs
  const int split = job & (SPLIT_ - 1);
  const int jid = job >> 2;                    // 0..1023
  const int h = jid >> 6;
  const int i0 = (jid & 63) << 4;
  const int jbeg = split * JCHUNK_;
  const int ln = lane & 15, half = lane >> 4;
  const float scale = 0.14433756729740644f;  // 1/sqrt(48)

  v16bf aq0 = load_a_frag(q, CS_, i0, h * HD_, 32);
  v16bf aq1 = load_a_frag(q, CS_, i0, h * HD_ + 32, 16);
  v8f o0 = {}, o1 = {}, o2 = {};
  float mrow[8], lrow[8];
#pragma unroll
  for (int r = 0; r < 8; ++r) { mrow[r] = -1e30f; lrow[r] = 0.f; }
  __bf16* pt = lds_p[wave];

  for (int j0 = jbeg; j0 < jbeg + JCHUNK_; j0 += 32) {
    v16bf b00 = load_b_frag(k, CS_, j0,      h * HD_,      32);
    v16bf b01 = load_b_frag(k, CS_, j0,      h * HD_ + 32, 16);
    v16bf b10 = load_b_frag(k, CS_, j0 + 16, h * HD_,      32);
    v16bf b11 = load_b_frag(k, CS_, j0 + 16, h * HD_ + 32, 16);
    v8f s0 = {}, s1 = {};
    s0 = wmma_bf16(aq0, b00, s0); s0 = wmma_bf16(aq1, b01, s0);
    s1 = wmma_bf16(aq0, b10, s1); s1 = wmma_bf16(aq1, b11, s1);

    const float mb0 = (1.0f - mask[j0 + ln]) * (-1e6f);
    const float mb1 = (1.0f - mask[j0 + 16 + ln]) * (-1e6f);
    const float* zrow = zb + (size_t)h * NN_ * NN_ + (size_t)i0 * NN_ + j0;
#pragma unroll
    for (int r = 0; r < 8; ++r) {
      int mm = r + half * 8;
      float sv0 = s0[r] * scale + zrow[(size_t)mm * NN_ + ln] + mb0;
      float sv1 = s1[r] * scale + zrow[(size_t)mm * NN_ + 16 + ln] + mb1;
      float rmx = fmaxf(sv0, sv1);
#pragma unroll
      for (int off = 1; off < 16; off <<= 1)
        rmx = fmaxf(rmx, __shfl_xor(rmx, off, 32));
      float Mn = fmaxf(mrow[r], rmx);
      float corr = __expf(mrow[r] - Mn);
      float p0 = __expf(sv0 - Mn);
      float p1 = __expf(sv1 - Mn);
      float rsum = p0 + p1;
#pragma unroll
      for (int off = 1; off < 16; off <<= 1)
        rsum += __shfl_xor(rsum, off, 32);
      mrow[r] = Mn;
      lrow[r] = lrow[r] * corr + rsum;
      o0[r] *= corr; o1[r] *= corr; o2[r] *= corr;
      pt[mm * 32 + ln]      = (__bf16)p0;
      pt[mm * 32 + 16 + ln] = (__bf16)p1;
    }
    // P (16x32) re-layout C->A through wave-private LDS, then P @ V
    v16bf pa = load_a_frag(pt, 32, 0, 0, 32);
    v16bf bv0 = load_b_frag(vT, NN_, h * HD_,      j0, 32);
    v16bf bv1 = load_b_frag(vT, NN_, h * HD_ + 16, j0, 32);
    v16bf bv2 = load_b_frag(vT, NN_, h * HD_ + 32, j0, 32);
    o0 = wmma_bf16(pa, bv0, o0);
    o1 = wmma_bf16(pa, bv1, o1);
    o2 = wmma_bf16(pa, bv2, o2);
  }
  // unnormalized partial O + per-row (m, l)
  float* pos = po + (size_t)split * NN_ * CS_;
#pragma unroll
  for (int r = 0; r < 8; ++r) {
    int mm = r + half * 8;
    size_t rowoff = (size_t)(i0 + mm) * CS_ + h * HD_;
    pos[rowoff + ln]      = o0[r];
    pos[rowoff + 16 + ln] = o1[r];
    pos[rowoff + 32 + ln] = o2[r];
    if (lane == 0)
      pml[((size_t)(split * H_ + h)) * NN_ + (i0 + r)] =
          make_float2(mrow[r], lrow[r]);
    if (lane == 16)
      pml[((size_t)(split * H_ + h)) * NN_ + (i0 + 8 + r)] =
          make_float2(mrow[r], lrow[r]);
  }
}

// ---------------------------------------------------------------------------
// Combine split partials: softmax-rescale, normalize, sigmoid-gate -> bf16
// ---------------------------------------------------------------------------
__global__ void __launch_bounds__(256)
attn_combine_kernel(const float* __restrict__ po, const float2* __restrict__ pml,
                    const float* __restrict__ g, __bf16* __restrict__ go) {
  int idx = blockIdx.x * 256 + threadIdx.x;  // over N * C_S
  if (idx >= NN_ * CS_) return;
  int i = idx / CS_, col = idx - i * CS_;
  int h = col / HD_;
  float2 ml[SPLIT_];
  float M = -1e30f;
#pragma unroll
  for (int s = 0; s < SPLIT_; ++s) {
    ml[s] = pml[((size_t)(s * H_ + h)) * NN_ + i];
    M = fmaxf(M, ml[s].x);
  }
  float l = 0.f, acc = 0.f;
#pragma unroll
  for (int s = 0; s < SPLIT_; ++s) {
    float w = __expf(ml[s].x - M);
    l += ml[s].y * w;
    acc += po[(size_t)s * NN_ * CS_ + idx] * w;
  }
  go[idx] = (__bf16)(acc / l * g[idx]);
}

// ---------------------------------------------------------------------------
// Launch
// ---------------------------------------------------------------------------
extern "C" void kernel_launch(void* const* d_in, const int* in_sizes, int n_in,
                              void* d_out, int out_size, void* d_ws, size_t ws_size,
                              hipStream_t stream) {
  const float* s      = (const float*)d_in[0];
  const float* z      = (const float*)d_in[1];
  const float* mask   = (const float*)d_in[2];
  const float* ln_s_w = (const float*)d_in[3];
  const float* ln_s_b = (const float*)d_in[4];
  const float* Wq     = (const float*)d_in[5];
  const float* bq     = (const float*)d_in[6];
  const float* Wk     = (const float*)d_in[7];
  const float* Wv     = (const float*)d_in[8];
  const float* Wg     = (const float*)d_in[9];
  const float* ln_z_w = (const float*)d_in[10];
  const float* ln_z_b = (const float*)d_in[11];
  const float* Wz     = (const float*)d_in[12];
  const float* Wo     = (const float*)d_in[13];
  float* out = (float*)d_out;

  char* p = (char*)d_ws;
  auto carve = [&](size_t bytes) -> char* {
    char* r = p;
    p += (bytes + 255) & ~(size_t)255;
    return r;
  };
  const size_t NCS = (size_t)NN_ * CS_;
  const size_t WSZ = (size_t)CS_ * CS_;
  __bf16* sln = (__bf16*)carve(NCS * 2);
  __bf16* Wqb = (__bf16*)carve(WSZ * 2);
  __bf16* Wkb = (__bf16*)carve(WSZ * 2);
  __bf16* Wvb = (__bf16*)carve(WSZ * 2);
  __bf16* Wgb = (__bf16*)carve(WSZ * 2);
  __bf16* Wob = (__bf16*)carve(WSZ * 2);
  __bf16* Wzb = (__bf16*)carve((size_t)H_ * CZ_ * 2);
  __bf16* qb  = (__bf16*)carve(NCS * 2);
  __bf16* kb  = (__bf16*)carve(NCS * 2);
  __bf16* vTb = (__bf16*)carve(NCS * 2);
  float*  gf  = (float*)carve(NCS * 4);
  __bf16* gob = (__bf16*)carve(NCS * 2);
  float*  pof = (float*)carve((size_t)SPLIT_ * NCS * 4);
  float2* pml = (float2*)carve((size_t)SPLIT_ * H_ * NN_ * 8);
  float*  zbw = (float*)carve((size_t)H_ * NN_ * NN_ * 4);

  // 1) weight conversions
  int wn = (int)WSZ;
  cvt_f32_bf16<<<(wn + 255) / 256, 256, 0, stream>>>(Wq, Wqb, wn);
  cvt_f32_bf16<<<(wn + 255) / 256, 256, 0, stream>>>(Wk, Wkb, wn);
  cvt_f32_bf16<<<(wn + 255) / 256, 256, 0, stream>>>(Wv, Wvb, wn);
  cvt_f32_bf16<<<(wn + 255) / 256, 256, 0, stream>>>(Wg, Wgb, wn);
  cvt_f32_bf16<<<(wn + 255) / 256, 256, 0, stream>>>(Wo, Wob, wn);
  cvt_f32_bf16<<<(H_ * CZ_ + 255) / 256, 256, 0, stream>>>(Wz, Wzb, H_ * CZ_);

  // 2) layernorm(s) -> bf16
  ln_s_kernel<<<NN_, 256, 0, stream>>>(s, ln_s_w, ln_s_b, sln);

  // 3) projections (3072 wave-tiles -> 768 blocks of 4 waves each)
  const int gemm_blocks = (NN_ / 16) * (CS_ / 16) / 4;
  gemm_xwT<<<gemm_blocks, 128, 0, stream>>>(sln, Wqb, bq, nullptr, qb,
                                            NN_, CS_, CS_, /*bias*/1);
  gemm_xwT<<<gemm_blocks, 128, 0, stream>>>(sln, Wkb, nullptr, nullptr, kb,
                                            NN_, CS_, CS_, 0);
  gemm_xwT<<<gemm_blocks, 128, 0, stream>>>(sln, Wvb, nullptr, nullptr, vTb,
                                            NN_, CS_, CS_, /*trans*/8);
  gemm_xwT<<<gemm_blocks, 128, 0, stream>>>(sln, Wgb, nullptr, gf, nullptr,
                                            NN_, CS_, CS_, /*sigmoid|f32*/6);

  // 4) pair bias (1024 * 16 wave-jobs -> 4096 blocks), roofline-critical
  zb_kernel<<<(NN_ * 16) / 4, 128, 0, stream>>>(z, ln_z_w, ln_z_b, Wzb, zbw);

  // 5) attention partials: SPLIT_*1024 wave-jobs -> 1024 blocks
  attn_partial_kernel<<<(SPLIT_ * H_ * (NN_ / 16)) / 4, 128, 0, stream>>>(
      qb, kb, vTb, zbw, mask, pof, pml);

  // 5b) combine + gate
  attn_combine_kernel<<<(int)(NCS / 256), 256, 0, stream>>>(pof, pml, gf, gob);

  // 6) output projection -> fp32 d_out
  gemm_xwT<<<gemm_blocks, 128, 0, stream>>>(gob, Wob, nullptr, out, nullptr,
                                            NN_, CS_, CS_, /*f32*/4);

  (void)in_sizes; (void)n_in; (void)out_size; (void)ws_size;
}
